// LinearRNN_85023172591791
// MI455X (gfx1250) — matrix operand order repaired and checked by reading, not compiled, not run on hardware
//
#include <hip/hip_runtime.h>
#include <hip/hip_bf16.h>
#include <stdint.h>

// ---------------------------------------------------------------------------
// LinearRNN for MI455X (gfx1250): 4-layer linear RNN, B=32,T=1024,I=H=512,O=256
//
//  * bf16 WMMA (v_wmma_f32_16x16x32_bf16) with fp32 accumulators.
//  * pack_weights: W fp32 -> bf16 in exact per-lane WMMA B-fragment layout.
//  * rnn_pipeline: persistent 4-WG kernel (one WG = one layer = one WGP),
//    pipelined over time via global ring buffers + acquire/release counters.
//    Step input staged once into LDS; B fragments double-buffered in regs
//    with sched_barriers pinning the prefetch a full k-tile ahead of use.
//  * readout: parallel WMMA GEMM  y = h @ W_ro^T + b_ro.
//
// Workspace (dwords):   ~5.25 MB of d_ws required
//   [0       , 524288)  packed W_ih   [524288 ,1048576) packed W_hh
//   [1048576 ,1114112)  packed W_ro   [1114112,1310720) rings (3*8 slots bf16)
//   [1310720 ,1310728)  counters prod[4], cons[4]
// ---------------------------------------------------------------------------

#define B_    32
#define T_    1024
#define I_    512
#define H_    512
#define O_    256
#define RINGN 8

#define WIH_DW  0
#define WHH_DW  524288
#define WRO_DW  1048576
#define RING_DW 1114112
#define CNT_DW  1310720

#if defined(__has_builtin) && __has_builtin(__builtin_amdgcn_sched_barrier)
#define SCHED_FENCE() __builtin_amdgcn_sched_barrier(0)
#else
#define SCHED_FENCE() ((void)0)
#endif

typedef __bf16        bf16x16 __attribute__((ext_vector_type(16)));
typedef float         f32x8   __attribute__((ext_vector_type(8)));
typedef float         f32x4   __attribute__((ext_vector_type(4), may_alias));
typedef unsigned int  u32x4   __attribute__((ext_vector_type(4), may_alias));
typedef unsigned int  u32x2   __attribute__((ext_vector_type(2), may_alias));

union Frag { bf16x16 v; u32x4 q[2]; };

__device__ __forceinline__ unsigned f2bf(float f) {
  union { float f; unsigned u; } v; v.f = f;
  unsigned r = v.u + 0x7FFFu + ((v.u >> 16) & 1u);   // round-to-nearest-even
  return r >> 16;
}

__device__ __forceinline__ f32x8 wmma_bf16(bf16x16 a, bf16x16 b, f32x8 c) {
  return __builtin_amdgcn_wmma_f32_16x16x32_bf16(false, a, false, b,
                                                 (short)0, c, false, false);
}

// A fragment (16x32 bf16) gathered from a row-major [32][512] bf16 buffer.
__device__ __forceinline__ Frag load_a_bf16(const unsigned short* base,
                                            int m0, int k0, int lane) {
  Frag f;
  int row = m0 + (lane & 15);
  int sel = (lane >> 4) * 8;
  const unsigned short* p = base + row * H_ + k0 + sel;
  f.q[0] = *(const u32x4*)(p);
  f.q[1] = *(const u32x4*)(p + 16);
  return f;
}

// B fragment from packed buffer: 32B contiguous per lane.
__device__ __forceinline__ Frag load_b(const unsigned* packBase, int fragIdx, int lane) {
  const u32x4* p = (const u32x4*)(packBase + (size_t)fragIdx * 256 + lane * 8);
  Frag f; f.q[0] = p[0]; f.q[1] = p[1]; return f;
}

// ---------------------------------------------------------------------------
// Kernel 1: pack weights; zero counters.  Grid 544x256 (4352 frags * 32 lanes)
// ---------------------------------------------------------------------------
__global__ __launch_bounds__(256) void pack_weights(
    const float* __restrict__ Wih, const float* __restrict__ Whh,
    const float* __restrict__ Wro, unsigned* __restrict__ ws32) {
  unsigned g = blockIdx.x * blockDim.x + threadIdx.x;
  if (g < 8) ((int*)(ws32 + CNT_DW))[g] = 0;
  unsigned frag = g >> 5, lane = g & 31;
  if (frag >= 4352) return;

  const float* src; unsigned* dst; unsigned K, nt, kt;
  if (frag < 2048) {                 // W_ih: frag = l*512 + nt*16 + kt
    unsigned l = frag >> 9, r = frag & 511;
    nt = r >> 4; kt = r & 15;
    src = Wih + (size_t)l * H_ * I_;
    dst = ws32 + WIH_DW + (size_t)frag * 256;
    K = I_;
  } else if (frag < 4096) {          // W_hh
    unsigned f = frag - 2048;
    unsigned l = f >> 9, r = f & 511;
    nt = r >> 4; kt = r & 15;
    src = Whh + (size_t)l * H_ * H_;
    dst = ws32 + WHH_DW + (size_t)f * 256;
    K = H_;
  } else {                           // W_ro: frag = nt*16 + kt
    unsigned f = frag - 4096;
    nt = f >> 4; kt = f & 15;
    src = Wro;
    dst = ws32 + WRO_DW + (size_t)f * 256;
    K = H_;
  }
  unsigned n  = nt * 16 + (lane & 15);
  unsigned kb = kt * 32 + (lane >> 4) * 8;
  const float* row = src + (size_t)n * K + kb;   // B[k][n] = W[n][k]
  unsigned* out = dst + lane * 8;
#pragma unroll
  for (int j = 0; j < 8; ++j)
    out[j] = f2bf(row[2 * j + (j >= 4 ? 8 : 0)]) | (f2bf(row[2 * j + 1 + (j >= 4 ? 8 : 0)]) << 16);
}

// ---------------------------------------------------------------------------
// RNN inner-loop helpers: B-fragment set ping-pong + one k-tile of compute.
// ---------------------------------------------------------------------------
struct BSet { Frag ih[4]; Frag hh[4]; };

__device__ __forceinline__ void load_bset(BSet& b, const unsigned* WihP,
                                          const unsigned* WhhP, int wave,
                                          int kt, int lane) {
#pragma unroll
  for (int n = 0; n < 4; ++n) {
    const int fi = (wave * 4 + n) * 16 + kt;
    b.ih[n] = load_b(WihP, fi, lane);
    b.hh[n] = load_b(WhhP, fi, lane);
  }
}

__device__ __forceinline__ void compute_kt(f32x8 acc[2][4], const BSet& b,
                                           const unsigned short* bufIn,
                                           const unsigned short* hPrev,
                                           int k0, int lane) {
  Frag aih[2], ahh[2];
#pragma unroll
  for (int m = 0; m < 2; ++m) {
    aih[m] = load_a_bf16(bufIn,  m * 16, k0, lane);   // ds_load_b128
    ahh[m] = load_a_bf16(hPrev,  m * 16, k0, lane);
  }
#pragma unroll
  for (int n = 0; n < 4; ++n) {
#pragma unroll
    for (int m = 0; m < 2; ++m) {
      acc[m][n] = wmma_bf16(aih[m].v, b.ih[n].v, acc[m][n]);
      acc[m][n] = wmma_bf16(ahh[m].v, b.hh[n].v, acc[m][n]);
    }
  }
}

// ---------------------------------------------------------------------------
// Kernel 2: persistent pipelined RNN.  Grid = 4 WGs (one per layer), 256 thr.
// LDS: h double buffer (64KB) + staged step input (32KB) = 96KB of 320KB.
// ---------------------------------------------------------------------------
__global__ __launch_bounds__(256, 1) void rnn_pipeline(
    const float* __restrict__ x, const float* __restrict__ h0,
    const float* __restrict__ bias, float* __restrict__ out_h,
    unsigned* __restrict__ ws32) {
  const int l    = blockIdx.x;
  const int tid  = threadIdx.x;
  const int wave = tid >> 5;             // wave32
  const int lane = tid & 31;

  const unsigned* WihP = ws32 + WIH_DW + (size_t)l * 512 * 256;
  const unsigned* WhhP = ws32 + WHH_DW + (size_t)l * 512 * 256;
  unsigned short* ringAll = (unsigned short*)(ws32 + RING_DW);
  int* prod = (int*)(ws32 + CNT_DW);
  int* cons = prod + 4;

  __shared__ __align__(16) unsigned short hbuf[2][B_ * H_];   // 64 KB
  __shared__ __align__(16) unsigned short bufIn[B_ * H_];     // 32 KB

  { // h0 -> LDS bf16
    const float* h0l = h0 + (size_t)l * B_ * H_;
    for (int i = tid; i < B_ * H_; i += 256)
      hbuf[0][i] = (unsigned short)f2bf(h0l[i]);
  }
  __syncthreads();

  const int n0 = wave * 64;
  float bv[4];
#pragma unroll
  for (int n = 0; n < 4; ++n)
    bv[n] = bias[l * H_ + n0 + n * 16 + (lane & 15)];

  unsigned short* ringIn  = (l > 0) ? ringAll + (size_t)(l - 1) * RINGN * B_ * H_ : nullptr;
  unsigned short* ringOut = (l < 3) ? ringAll + (size_t)l       * RINGN * B_ * H_ : nullptr;

  int parity = 0;
  for (int t = 0; t < T_; ++t) {
    // --- pipeline waits ---
    if (tid == 0) {
      if (l > 0)
        while (__hip_atomic_load(&prod[l - 1], __ATOMIC_ACQUIRE,
                                 __HIP_MEMORY_SCOPE_AGENT) <= t)
          __builtin_amdgcn_s_sleep(1);
      if (l < 3 && t >= RINGN)
        while (__hip_atomic_load(&cons[l], __ATOMIC_ACQUIRE,
                                 __HIP_MEMORY_SCOPE_AGENT) <= t - RINGN)
          __builtin_amdgcn_s_sleep(1);
    }
    __syncthreads();

    // --- kick off the first B-fragment prefetch; it flies during staging ---
    BSet b0, b1;
    load_bset(b0, WihP, WhhP, wave, 0, lane);
    SCHED_FENCE();

    // --- stage this step's input into LDS (once per WG, coalesced) ---
    if (l == 0) {            // x_t fp32 -> bf16
      for (int i = tid; i < (B_ * I_) / 4; i += 256) {
        const int row = i >> 7;                 // 128 float4 per row
        const int c4  = (i & 127) << 2;
        const f32x4 v = *(const f32x4*)(x + ((size_t)row * T_ + t) * I_ + c4);
        u32x2 p; p[0] = f2bf(v[0]) | (f2bf(v[1]) << 16);
        p[1] = f2bf(v[2]) | (f2bf(v[3]) << 16);
        *(u32x2*)(bufIn + row * I_ + c4) = p;
      }
    } else {                 // ring slot bf16 -> LDS
      const unsigned short* slotIn = ringIn + (size_t)(t & (RINGN - 1)) * B_ * H_;
      const u32x4* s = (const u32x4*)slotIn;
      u32x4* d = (u32x4*)bufIn;
      for (int i = tid; i < (B_ * H_) / 8; i += 256) d[i] = s[i];
    }
    __syncthreads();

    const unsigned short* hPrev = hbuf[parity];
    unsigned short*       hNew  = hbuf[parity ^ 1];

    f32x8 acc[2][4];
#pragma unroll
    for (int m = 0; m < 2; ++m)
#pragma unroll
      for (int n = 0; n < 4; ++n) {
        f32x8 tmp;
#pragma unroll
        for (int r = 0; r < 8; ++r) tmp[r] = bv[n];
        acc[m][n] = tmp;
      }

    // --- K loop: ping-pong B fragments; sched_barriers pin the prefetch
    //     phase a full k-tile (16 WMMAs) ahead of its consumers.
#pragma unroll
    for (int kt = 0; kt < 16; kt += 2) {
      load_bset(b1, WihP, WhhP, wave, kt + 1, lane);
      SCHED_FENCE();
      compute_kt(acc, b0, bufIn, hPrev, kt * 32, lane);
      SCHED_FENCE();
      if (kt + 2 < 16) load_bset(b0, WihP, WhhP, wave, kt + 2, lane);
      SCHED_FENCE();
      compute_kt(acc, b1, bufIn, hPrev, (kt + 1) * 32, lane);
      SCHED_FENCE();
    }

    // --- write back h (C/D layout: lanes 0-15 -> M=r, lanes 16-31 -> M=r+8)
    const int rb = (lane >> 4) * 8;
#pragma unroll
    for (int m = 0; m < 2; ++m)
#pragma unroll
      for (int n = 0; n < 4; ++n) {
        const int col = n0 + n * 16 + (lane & 15);
#pragma unroll
        for (int r = 0; r < 8; ++r) {
          const int row = m * 16 + rb + r;
          const float v = acc[m][n][r];
          hNew[row * H_ + col] = (unsigned short)f2bf(v);
          if (l == 3)
            out_h[((size_t)row * T_ + t) * H_ + col] = v;   // h_last_seq
        }
      }
    __syncthreads();

    if (l < 3) {   // publish slot: LDS -> global ring
      unsigned short* slotOut = ringOut + (size_t)(t & (RINGN - 1)) * B_ * H_;
      const u32x4* s = (const u32x4*)hNew;
      u32x4*       d = (u32x4*)slotOut;
      for (int i = tid; i < (B_ * H_) / 8; i += 256) d[i] = s[i];
    }
    __threadfence();
    __syncthreads();
    if (tid == 0) {
      if (l < 3) __hip_atomic_store(&prod[l], t + 1, __ATOMIC_RELEASE,
                                    __HIP_MEMORY_SCOPE_AGENT);
      if (l > 0) __hip_atomic_store(&cons[l - 1], t + 1, __ATOMIC_RELEASE,
                                    __HIP_MEMORY_SCOPE_AGENT);
    }
    parity ^= 1;
  }
}

// ---------------------------------------------------------------------------
// Kernel 3: readout GEMM  y[32768,256] = h[32768,512] @ W_ro^T + b_ro.
// 1024 WGs x 128 thr; 32x512 h panel staged in LDS bf16; B frags ping-ponged.
// ---------------------------------------------------------------------------
__global__ __launch_bounds__(128, 1) void readout(
    const float* __restrict__ hseq, const float* __restrict__ bro,
    float* __restrict__ y, const unsigned* __restrict__ ws32) {
  const unsigned* WroP = ws32 + WRO_DW;
  const int wg   = blockIdx.x;
  const int tid  = threadIdx.x;
  const int wave = tid >> 5;
  const int lane = tid & 31;

  __shared__ __align__(16) unsigned short panel[32 * H_];   // 32 KB

  Frag c0[4], c1[4];
#pragma unroll
  for (int n = 0; n < 4; ++n) c0[n] = load_b(WroP, (wave * 4 + n) * 16 + 0, lane);
  SCHED_FENCE();

  {
    const float* src = hseq + (size_t)wg * 32 * H_;
    for (int i = tid; i < (32 * H_) / 4; i += 128) {
      const f32x4 v = *(const f32x4*)(src + i * 4);
      u32x2 p; p[0] = f2bf(v[0]) | (f2bf(v[1]) << 16);
      p[1] = f2bf(v[2]) | (f2bf(v[3]) << 16);
      *(u32x2*)(panel + i * 4) = p;
    }
  }
  __syncthreads();

  const int n0 = wave * 64;
  float bv[4];
#pragma unroll
  for (int n = 0; n < 4; ++n) bv[n] = bro[n0 + n * 16 + (lane & 15)];

  f32x8 acc[2][4];
#pragma unroll
  for (int m = 0; m < 2; ++m)
#pragma unroll
    for (int n = 0; n < 4; ++n) {
      f32x8 tmp;
#pragma unroll
      for (int r = 0; r < 8; ++r) tmp[r] = bv[n];
      acc[m][n] = tmp;
    }

#pragma unroll
  for (int kt = 0; kt < 16; kt += 2) {
#pragma unroll
    for (int n = 0; n < 4; ++n)
      c1[n] = load_b(WroP, (wave * 4 + n) * 16 + kt + 1, lane);
    SCHED_FENCE();
    {
      Frag a[2];
#pragma unroll
      for (int m = 0; m < 2; ++m) a[m] = load_a_bf16(panel, m * 16, kt * 32, lane);
#pragma unroll
      for (int n = 0; n < 4; ++n) {
        acc[0][n] = wmma_bf16(a[0].v, c0[n].v, acc[0][n]);
        acc[1][n] = wmma_bf16(a[1].v, c0[n].v, acc[1][n]);
      }
    }
    SCHED_FENCE();
    if (kt + 2 < 16)
#pragma unroll
      for (int n = 0; n < 4; ++n)
        c0[n] = load_b(WroP, (wave * 4 + n) * 16 + kt + 2, lane);
    SCHED_FENCE();
    {
      Frag a[2];
#pragma unroll
      for (int m = 0; m < 2; ++m)
        a[m] = load_a_bf16(panel, m * 16, (kt + 1) * 32, lane);
#pragma unroll
      for (int n = 0; n < 4; ++n) {
        acc[0][n] = wmma_bf16(a[0].v, c1[n].v, acc[0][n]);
        acc[1][n] = wmma_bf16(a[1].v, c1[n].v, acc[1][n]);
      }
    }
    SCHED_FENCE();
  }

  const int rb = (lane >> 4) * 8;
#pragma unroll
  for (int m = 0; m < 2; ++m)
#pragma unroll
    for (int n = 0; n < 4; ++n) {
      const int col = n0 + n * 16 + (lane & 15);
#pragma unroll
      for (int r = 0; r < 8; ++r) {
        const int row = wg * 32 + m * 16 + rb + r;
        y[(size_t)row * O_ + col] = acc[m][n][r];
      }
    }
}

// ---------------------------------------------------------------------------
extern "C" void kernel_launch(void* const* d_in, const int* in_sizes, int n_in,
                              void* d_out, int out_size, void* d_ws, size_t ws_size,
                              hipStream_t stream) {
  (void)in_sizes; (void)n_in; (void)out_size; (void)ws_size;
  const float* x    = (const float*)d_in[0];   // (B,T,I)
  const float* h0   = (const float*)d_in[1];   // (L,B,H)
  const float* Wih  = (const float*)d_in[2];   // (L,H,I)
  const float* Whh  = (const float*)d_in[3];   // (L,H,H)
  const float* bias = (const float*)d_in[4];   // (L,H)
  const float* Wro  = (const float*)d_in[5];   // (O,H)
  const float* bro  = (const float*)d_in[6];   // (O,)

  float* y    = (float*)d_out;                    // (B,T,O)
  float* hseq = y + (size_t)B_ * T_ * O_;         // (B,T,H)
  unsigned* ws32 = (unsigned*)d_ws;

  pack_weights<<<544, 256, 0, stream>>>(Wih, Whh, Wro, ws32);
  rnn_pipeline<<<4, 256, 0, stream>>>(x, h0, bias, hseq, ws32);
  readout<<<1024, 128, 0, stream>>>(hseq, bro, y, ws32);
}
